// MultiHeadedAttention_1623497638203
// MI455X (gfx1250) — matrix operand order repaired and checked by reading, not compiled, not run on hardware
//
#include <hip/hip_runtime.h>
#include <hip/hip_bf16.h>

// ---------------------------------------------------------------------------
// MHA forward for B=4, S=2048, H=768, NH=12, PH=64 on gfx1250 (wave32, WMMA).
//   1) q/k/v = x @ W^T + b   (WMMA f16 GEMM, f32 accum; V stored transposed)
//   2) flash attention, K/V blocks double-buffered via async-to-LDS loads
//   3) out = attn @ Wo^T + bo (WMMA f16 GEMM, f32 out, NT stores)
// ---------------------------------------------------------------------------

#define BB   4
#define SS   2048
#define HH   768
#define NHD  12
#define PHD  64
#define BS   (BB * SS)           // 8192 rows

typedef __attribute__((ext_vector_type(16))) _Float16 v16h;
typedef __attribute__((ext_vector_type(8)))  _Float16 v8h;
typedef __attribute__((ext_vector_type(8)))  float    v8f;

__device__ __forceinline__ v8f wmma32(v16h a, v16h b, v8f c) {
    // D = A(16x32 f16) * B(32x16 f16) + C(16x16 f32)
    return __builtin_amdgcn_wmma_f32_16x16x32_f16(
        false, a, false, b, (short)0, c, false, false);
}

// ---- CDNA5 async global->LDS copy (ASYNCcnt) ------------------------------
// GLOBAL_LOAD_ASYNC_TO_LDS_B128: VDST = LDS byte address, VADDR = 64-bit GVA.
__device__ __forceinline__ void async_b128(const void* g, void* l) {
    asm volatile("global_load_async_to_lds_b128 %0, %1, off"
                 :: "v"((unsigned int)(unsigned long long)l), "v"(g)
                 : "memory");
}
__device__ __forceinline__ void wait_async0() {
    asm volatile("s_wait_asynccnt 0x0" ::: "memory");
}

// A fragment (16x32, MxK). p at tile base, ld in halves.
// lanes 0-15: e0..7 -> K=e, e8..15 -> K=16+(e-8);  lanes 16-31: +8.
__device__ __forceinline__ v16h load_frag_a(const _Float16* p, int ld, int lane) {
    const int mr = lane & 15, hh = lane >> 4;
    union { v16h v; v8h h[2]; } f;
    const _Float16* r = p + mr * ld + 8 * hh;
    f.h[0] = *(const v8h*)(r);
    f.h[1] = *(const v8h*)(r + 16);
    return f.v;
}

// B fragment (32x16, KxN) sourced from row-major B^T rows ("[n][k]").
// lane holds column n = lane&15; e0..7 -> K=e+8*(lane>>4), e8..15 -> K=16+(e-8)+8*(lane>>4).
__device__ __forceinline__ v16h load_frag_b(const _Float16* p, int ld, int lane) {
    const int mr = lane & 15, hh = lane >> 4;
    union { v16h v; v8h h[2]; } f;
    const _Float16* r = p + mr * ld + 8 * hh;
    f.h[0] = *(const v8h*)(r);
    f.h[1] = *(const v8h*)(r + 16);
    return f.v;
}

// 16 consecutive f32 -> 16 f16 into LDS
__device__ __forceinline__ void stage16(const float* __restrict__ g, _Float16* l) {
    const float4* s = (const float4*)g;
    float4 f0 = s[0], f1 = s[1], f2 = s[2], f3 = s[3];
    l[0]  = (_Float16)f0.x; l[1]  = (_Float16)f0.y;
    l[2]  = (_Float16)f0.z; l[3]  = (_Float16)f0.w;
    l[4]  = (_Float16)f1.x; l[5]  = (_Float16)f1.y;
    l[6]  = (_Float16)f1.z; l[7]  = (_Float16)f1.w;
    l[8]  = (_Float16)f2.x; l[9]  = (_Float16)f2.y;
    l[10] = (_Float16)f2.z; l[11] = (_Float16)f2.w;
    l[12] = (_Float16)f3.x; l[13] = (_Float16)f3.y;
    l[14] = (_Float16)f3.z; l[15] = (_Float16)f3.w;
}

// ---------------------------------------------------------------------------
// Kernel 1: y = x @ W^T + bias, f32 in, f16 out with head-split routing.
// Tile: 64 rows x 128 cols per 256-thread block; K staged 64 at a time.
// vmode 0: out[B,NH,S,PH]   vmode 1: out[B,NH,PH,S] (transposed, for V)
// ---------------------------------------------------------------------------
__global__ __launch_bounds__(256)
void qkv_proj_kernel(const float* __restrict__ x, const float* __restrict__ W,
                     const float* __restrict__ bias, _Float16* __restrict__ out,
                     int vmode) {
    __shared__ __align__(16) _Float16 sA[64 * 64];    //  8 KB
    __shared__ __align__(16) _Float16 sB[128 * 64];   // 16 KB

    const int tid  = threadIdx.x;
    const int lane = tid & 31, wave = tid >> 5;
    const int r0 = blockIdx.x * 64;
    const int c0 = blockIdx.y * 128;
    const int mbase = (wave & 3) * 16;
    const int nbase = (wave >> 2) * 64;

    const int arow = tid >> 2, aseg = tid & 3;   // A loader: 16 floats each
    const int brow = tid >> 1, bseg = tid & 1;   // B loader: 32 floats each

    v8f acc[4] = {};

    for (int kt = 0; kt < HH / 64; ++kt) {
        const int k0 = kt * 64;
        __syncthreads();
        stage16(x + (size_t)(r0 + arow) * HH + k0 + aseg * 16,
                &sA[arow * 64 + aseg * 16]);
        stage16(W + (size_t)(c0 + brow) * HH + k0 + bseg * 32,
                &sB[brow * 64 + bseg * 32]);
        stage16(W + (size_t)(c0 + brow) * HH + k0 + bseg * 32 + 16,
                &sB[brow * 64 + bseg * 32 + 16]);
        __syncthreads();

        for (int kk = 0; kk < 2; ++kk) {
            v16h a = load_frag_a(&sA[mbase * 64 + kk * 32], 64, lane);
            v16h bf[4];
            for (int n = 0; n < 4; ++n)
                bf[n] = load_frag_b(&sB[(nbase + n * 16) * 64 + kk * 32], 64, lane);
            for (int n = 0; n < 4; ++n)
                acc[n] = wmma32(a, bf[n], acc[n]);
        }
    }

    const int hh = lane >> 4, mr = lane & 15;
    for (int n = 0; n < 4; ++n) {
        const int gcol = c0 + nbase + n * 16 + mr;
        const float bv = bias[gcol];
        const int head = gcol >> 6, p = gcol & 63;
        for (int j = 0; j < 8; ++j) {
            const int grow = r0 + mbase + j + 8 * hh;
            const int bb = grow >> 11, s = grow & (SS - 1);
            const float val = acc[n][j] + bv;
            size_t idx;
            if (vmode == 0)  // [B,NH,S,PH]
                idx = ((size_t)(bb * NHD + head) * SS + s) * PHD + p;
            else             // [B,NH,PH,S]
                idx = ((size_t)(bb * NHD + head) * PHD + p) * SS + s;
            out[idx] = (_Float16)val;
        }
    }
}

// ---------------------------------------------------------------------------
// Kernel 2: flash attention. Block = (b, head, 128-row Q tile), 8 waves,
// each wave owns 16 Q rows x full 64-key block. K/V double-buffered with
// async-to-LDS loads; online softmax; P round-trips wave-private LDS.
// ---------------------------------------------------------------------------
__global__ __launch_bounds__(256)
void flash_attn_kernel(const _Float16* __restrict__ q16,
                       const _Float16* __restrict__ k16,
                       const _Float16* __restrict__ vT16,
                       const float* __restrict__ mask,
                       _Float16* __restrict__ attn16) {
    __shared__ __align__(16) _Float16 sQ[128 * 64];      // 16 KB
    __shared__ __align__(16) _Float16 sK[2][64 * 64];    // 16 KB
    __shared__ __align__(16) _Float16 sV[2][64 * 64];    // 16 KB (Vt: [p][k])
    __shared__ __align__(16) _Float16 sP[8][16 * 64];    // 16 KB wave-private

    const int tid  = threadIdx.x;
    const int lane = tid & 31, wave = tid >> 5;
    const int q0   = blockIdx.x * 128;
    const int head = blockIdx.y;
    const int bb   = blockIdx.z;
    const size_t bh_off = ((size_t)bb * NHD + head) * (size_t)SS * PHD;
    const int mr = lane & 15, hh = lane >> 4;
    const int krow = tid >> 2, kq = tid & 3;

    auto stage_kv = [&](int kb2, int buf) {
        const int kbase2 = kb2 * 64;
        const _Float16* gk = k16 + bh_off + (size_t)(kbase2 + krow) * PHD + kq * 16;
        _Float16* lk = &sK[buf][krow * 64 + kq * 16];
        async_b128(gk, lk); async_b128(gk + 8, lk + 8);
        const _Float16* gv = vT16 + bh_off + (size_t)krow * SS + kbase2 + kq * 16;
        _Float16* lv = &sV[buf][krow * 64 + kq * 16];
        async_b128(gv, lv); async_b128(gv + 8, lv + 8);
    };

    {   // Q tile 128x64 halves via async copies
        const int row = tid >> 1, hf = tid & 1;
        const _Float16* g = q16 + bh_off + (size_t)(q0 + row) * PHD + hf * 32;
        _Float16* l = &sQ[row * 64 + hf * 32];
        async_b128(g,      l);      async_b128(g + 8,  l + 8);
        async_b128(g + 16, l + 16); async_b128(g + 24, l + 24);
    }
    stage_kv(0, 0);
    wait_async0();
    __syncthreads();

    v16h qf[2];
    qf[0] = load_frag_a(&sQ[(wave * 16) * 64 +  0], 64, lane);
    qf[1] = load_frag_a(&sQ[(wave * 16) * 64 + 32], 64, lane);

    v8f o[4] = {};
    float mrun[8], lrun[8];
    for (int j = 0; j < 8; ++j) { mrun[j] = -__builtin_inff(); lrun[j] = 0.f; }

    const float scale = 0.125f;               // 1/sqrt(PH)
    const float LOG2E = 1.4426950408889634f;
    const float* maskb = mask + (size_t)bb * SS * SS;

    for (int kb = 0; kb < SS / 64; ++kb) {
        const int cur = kb & 1;
        if (kb + 1 < SS / 64) stage_kv(kb + 1, cur ^ 1);   // prefetch next block
        const int kbase = kb * 64;

        // scores S = Q K^T * scale + mask (mask is stream-once -> NT loads)
        v8f sc[4] = {};
        for (int kk = 0; kk < 2; ++kk) {
            v16h kf[4];
            for (int n = 0; n < 4; ++n)
                kf[n] = load_frag_b(&sK[cur][(n * 16) * 64 + kk * 32], 64, lane);
            for (int n = 0; n < 4; ++n)
                sc[n] = wmma32(qf[kk], kf[n], sc[n]);
        }
        float s[4][8];
        for (int n = 0; n < 4; ++n) {
            const int kcol = kbase + n * 16 + mr;
            for (int j = 0; j < 8; ++j) {
                const int qrow = q0 + wave * 16 + j + 8 * hh;
                const float mval =
                    __builtin_nontemporal_load(&maskb[(size_t)qrow * SS + kcol]);
                s[n][j] = sc[n][j] * scale + mval;
            }
        }

        // online softmax (one row spans a 16-lane half-wave)
        for (int j = 0; j < 8; ++j) {
            float m = fmaxf(fmaxf(s[0][j], s[1][j]), fmaxf(s[2][j], s[3][j]));
            for (int off = 1; off < 16; off <<= 1)
                m = fmaxf(m, __shfl_xor(m, off, 32));
            const float mnew = fmaxf(mrun[j], m);
            const float corr = exp2f((mrun[j] - mnew) * LOG2E);
            float rs = 0.f;
            for (int n = 0; n < 4; ++n) {
                const float p = exp2f((s[n][j] - mnew) * LOG2E);
                s[n][j] = p;
                rs += p;
            }
            for (int off = 1; off < 16; off <<= 1)
                rs += __shfl_xor(rs, off, 32);
            lrun[j] = lrun[j] * corr + rs;
            mrun[j] = mnew;
            for (int n = 0; n < 4; ++n) o[n][j] *= corr;
        }

        // C-layout -> A-layout via wave-private LDS scratch
        _Float16* pw = &sP[wave][0];
        for (int n = 0; n < 4; ++n)
            for (int j = 0; j < 8; ++j)
                pw[(j + 8 * hh) * 64 + n * 16 + mr] = (_Float16)s[n][j];

        // O += P V
        for (int kk = 0; kk < 2; ++kk) {
            v16h pf = load_frag_a(&sP[wave][kk * 32], 64, lane);
            v16h vf[4];
            for (int n = 0; n < 4; ++n)
                vf[n] = load_frag_b(&sV[cur][(n * 16) * 64 + kk * 32], 64, lane);
            for (int n = 0; n < 4; ++n)
                o[n] = wmma32(pf, vf[n], o[n]);
        }

        wait_async0();      // my prefetch for kb+1 has landed
        __syncthreads();    // everyone done reading buf 'cur' + prefetch visible
    }

    // normalize and emit [B,S,H] f16
    for (int n = 0; n < 4; ++n) {
        const int col = head * PHD + n * 16 + mr;
        for (int j = 0; j < 8; ++j) {
            const int row = q0 + wave * 16 + j + 8 * hh;
            const float inv = 1.f / lrun[j];
            attn16[((size_t)bb * SS + row) * HH + col] = (_Float16)(o[n][j] * inv);
        }
    }
}

// ---------------------------------------------------------------------------
// Kernel 3: out = attn @ Wo^T + bo, f16 A via async-to-LDS, f32 NT output.
// ---------------------------------------------------------------------------
__global__ __launch_bounds__(256)
void out_proj_kernel(const _Float16* __restrict__ attn,
                     const float* __restrict__ W,
                     const float* __restrict__ bias,
                     float* __restrict__ out) {
    __shared__ __align__(16) _Float16 sA[64 * 64];    //  8 KB
    __shared__ __align__(16) _Float16 sB[128 * 64];   // 16 KB

    const int tid  = threadIdx.x;
    const int lane = tid & 31, wave = tid >> 5;
    const int r0 = blockIdx.x * 64;
    const int c0 = blockIdx.y * 128;
    const int mbase = (wave & 3) * 16;
    const int nbase = (wave >> 2) * 64;

    const int arow = tid >> 2, aseg = tid & 3;
    const int brow = tid >> 1, bseg = tid & 1;

    v8f acc[4] = {};

    for (int kt = 0; kt < HH / 64; ++kt) {
        const int k0 = kt * 64;
        __syncthreads();
        {   // A already f16: async global->LDS copy
            const _Float16* g = attn + (size_t)(r0 + arow) * HH + k0 + aseg * 16;
            _Float16* l = &sA[arow * 64 + aseg * 16];
            async_b128(g, l); async_b128(g + 8, l + 8);
        }
        stage16(W + (size_t)(c0 + brow) * HH + k0 + bseg * 32,
                &sB[brow * 64 + bseg * 32]);
        stage16(W + (size_t)(c0 + brow) * HH + k0 + bseg * 32 + 16,
                &sB[brow * 64 + bseg * 32 + 16]);
        wait_async0();
        __syncthreads();

        for (int kk = 0; kk < 2; ++kk) {
            v16h a = load_frag_a(&sA[mbase * 64 + kk * 32], 64, lane);
            v16h bf[4];
            for (int n = 0; n < 4; ++n)
                bf[n] = load_frag_b(&sB[(nbase + n * 16) * 64 + kk * 32], 64, lane);
            for (int n = 0; n < 4; ++n)
                acc[n] = wmma32(a, bf[n], acc[n]);
        }
    }

    const int hh = lane >> 4, mr = lane & 15;
    for (int n = 0; n < 4; ++n) {
        const int gcol = c0 + nbase + n * 16 + mr;
        const float bv = bias[gcol];
        for (int j = 0; j < 8; ++j) {
            const int grow = r0 + mbase + j + 8 * hh;
            __builtin_nontemporal_store(acc[n][j] + bv, &out[(size_t)grow * HH + gcol]);
        }
    }
}

// ---------------------------------------------------------------------------
extern "C" void kernel_launch(void* const* d_in, const int* in_sizes, int n_in,
                              void* d_out, int out_size, void* d_ws, size_t ws_size,
                              hipStream_t stream) {
    const float* key   = (const float*)d_in[0];
    const float* value = (const float*)d_in[1];
    const float* query = (const float*)d_in[2];
    const float* mask  = (const float*)d_in[3];
    const float* Wq    = (const float*)d_in[4];
    const float* bq    = (const float*)d_in[5];
    const float* Wk    = (const float*)d_in[6];
    const float* bk    = (const float*)d_in[7];
    const float* Wv    = (const float*)d_in[8];
    const float* bv    = (const float*)d_in[9];
    const float* Wo    = (const float*)d_in[10];
    const float* bo    = (const float*)d_in[11];
    float* out = (float*)d_out;

    const size_t QKV = (size_t)BB * NHD * SS * PHD;   // 6.29M halves each
    _Float16* q16    = (_Float16*)d_ws;
    _Float16* k16    = q16 + QKV;
    _Float16* vT16   = k16 + QKV;
    _Float16* attn16 = vT16 + QKV;

    dim3 gproj(BS / 64, HH / 128);
    qkv_proj_kernel<<<gproj, 256, 0, stream>>>(query, Wq, bq, q16, 0);
    qkv_proj_kernel<<<gproj, 256, 0, stream>>>(key,   Wk, bk, k16, 0);
    qkv_proj_kernel<<<gproj, 256, 0, stream>>>(value, Wv, bv, vT16, 1);

    dim3 gattn(SS / 128, NHD, BB);
    flash_attn_kernel<<<gattn, 256, 0, stream>>>(q16, k16, vT16, mask, attn16);

    out_proj_kernel<<<gproj, 256, 0, stream>>>(attn16, Wo, bo, out);
}
